// GNN_59863254171763
// MI455X (gfx1250) — matrix operand order repaired and checked by reading, not compiled, not run on hardware
//
#include <hip/hip_runtime.h>

#define HID 128
#define IN_FEAT 64
#define POS 3
#define TILE_E 64
#define TILE_N 64

typedef _Float16 half16 __attribute__((ext_vector_type(16)));
typedef _Float16 half4 __attribute__((ext_vector_type(4)));
typedef float float8 __attribute__((ext_vector_type(8)));

__device__ __forceinline__ float silu_f(float v) {
    return v / (1.0f + __expf(-v));
}

__device__ __forceinline__ float8 wmma32(half16 a, half16 b, float8 c) {
    // 8 args: (neg_a, A, neg_b, B, c_mod, C, reuse_a, reuse_b)
    return __builtin_amdgcn_wmma_f32_16x16x32_f16(false, a, false, b, (short)0, c,
                                                  false, false);
}

// A fragment: 16x32 f16 tile at (rowBase, kBase) of a row-major LDS buffer, leading dim lda.
// Layout per ISA: lanes 0-15 hold K=[0..7],[16..23]; lanes 16-31 hold K=[8..15],[24..31].
__device__ __forceinline__ half16 load_a(const _Float16* s, int rowBase, int kBase,
                                         int lane, int lda) {
    const int row = rowBase + (lane & 15);
    const int k0 = kBase + ((lane >> 4) << 3);
    const _Float16* p = s + row * lda + k0;
    half16 a;
#pragma unroll
    for (int e = 0; e < 8; ++e) a[e] = p[e];
#pragma unroll
    for (int e = 8; e < 16; ++e) a[e] = p[e + 8];
    return a;
}

// B fragment from pre-swizzled weights: each lane's 16 halves are contiguous (32 B),
// fragment-major layout: [kTile][nTile][lane][16]. Single half16 load -> 2x b128.
__device__ __forceinline__ half16 load_b_sw(const _Float16* wsw, int kTile, int nTile,
                                            int lane) {
    return *(const half16*)(wsw + (size_t)((((kTile << 3) + nTile) << 5) + lane) * 16);
}

// Swizzle-convert one row-major [K x HID] f32 weight matrix into the B-fragment layout.
// Output element o: e = o&15, lane = (o>>4)&31, nTile = (o>>9)&7, kTile = o>>12.
// Source: K = kTile*32 + (lane>=16 ? 8 : 0) + (e&7) + (e>=8 ? 16 : 0) ; N = nTile*16 + lane%16.
__global__ void swizzle_b_kernel(const float* __restrict__ src,
                                 _Float16* __restrict__ dst, int count) {
    int o = blockIdx.x * blockDim.x + threadIdx.x;
    if (o >= count) return;
    const int e = o & 15;
    const int lane = (o >> 4) & 31;
    const int nTile = (o >> 9) & 7;
    const int kTile = o >> 12;
    const int k = kTile * 32 + ((lane >> 4) << 3) + (e & 7) + ((e >> 3) << 4);
    const int n = nTile * 16 + (lane & 15);
    dst[o] = (_Float16)src[(size_t)k * HID + n];
}

__global__ void zero_kernel(float* __restrict__ p, int count) {
    int i = blockIdx.x * blockDim.x + threadIdx.x;
    if (i < count) p[i] = 0.0f;
}

// h = silu(concat(vel@ve_w + ve_b, x) @ node_w + node_b); one block per node, 128 threads.
__global__ __launch_bounds__(HID) void egnn_encoder_kernel(
    const float* __restrict__ vel, const float* __restrict__ x,
    const float* __restrict__ ve_w, const float* __restrict__ ve_b,
    const float* __restrict__ node_w, const float* __restrict__ node_b,
    float* __restrict__ h, _Float16* __restrict__ hf, int N) {
    const int nd = blockIdx.x;
    const int t = threadIdx.x;
    if (nd >= N) return;
    __shared__ float s_in[IN_FEAT + POS];
    if (t < IN_FEAT) {
        s_in[t] = vel[nd] * ve_w[t] + ve_b[t];
    } else if (t < IN_FEAT + POS) {
        s_in[t] = x[nd * POS + (t - IN_FEAT)];
    }
    __syncthreads();
    float acc = node_b[t];
#pragma unroll 1
    for (int k = 0; k < IN_FEAT + POS; ++k) acc += s_in[k] * node_w[k * HID + t];
    const float v = silu_f(acc);
    h[(size_t)nd * HID + t] = v;
    hf[(size_t)nd * HID + t] = (_Float16)v;
}

// Fused edge MLP: gather h[row],h[col] -> [E,257]x[257,128] -> SiLU -> x[128,128] -> SiLU
// -> atomic segment-sum into mi[row]. 64 edges / block, 8 waves; wave w owns N cols [16w,16w+16).
__global__ __launch_bounds__(256) void egnn_edge_kernel(
    const _Float16* __restrict__ hf, const int* __restrict__ ei,
    const float* __restrict__ charges,
    const _Float16* __restrict__ w1sw,       // swizzled [256 x 128] f16 fragments
    const float* __restrict__ w1_attr_row,   // fp32 row 256 of w1 (edge_attr weights)
    const float* __restrict__ b1,
    const _Float16* __restrict__ w2sw,       // swizzled [128 x 128] f16 fragments
    const float* __restrict__ b2,
    float* __restrict__ mi, int E) {
    __shared__ __align__(16) _Float16 s_hr[TILE_E * HID];
    __shared__ __align__(16) _Float16 s_hc[TILE_E * HID];
    __shared__ __align__(16) _Float16 s_mid[TILE_E * HID];
    __shared__ int s_r[TILE_E];
    __shared__ int s_c[TILE_E];
    __shared__ float s_attr[TILE_E];

    const int tid = threadIdx.x;
    const int e0 = blockIdx.x * TILE_E;

    if (tid < TILE_E) {
        const int e = e0 + tid;
        int r = -1, c = 0;
        float attr = 0.0f;
        if (e < E) {
            r = ei[e];
            c = ei[E + e];
            attr = charges[r] * charges[c];
        }
        s_r[tid] = r;
        s_c[tid] = c;
        s_attr[tid] = attr;
    }
    __syncthreads();

    // Cooperative gather: 16 bytes (8 halves) per thread-iteration.
    for (int idx = tid; idx < TILE_E * (HID / 8); idx += 256) {
        const int e = idx >> 4;
        const int ch = idx & 15;
        int r = s_r[e]; if (r < 0) r = 0;
        const int c = s_c[e];
        ((uint4*)s_hr)[idx] = ((const uint4*)(hf + (size_t)r * HID))[ch];
        ((uint4*)s_hc)[idx] = ((const uint4*)(hf + (size_t)c * HID))[ch];
    }
    __syncthreads();

    const int lane = tid & 31;
    const int nTile = tid >> 5;                 // 16-col slice per wave
    const int n = (nTile << 4) + (lane & 15);
    const int mSel = (lane >> 4) << 3;          // C/D: VGPR i -> M = i (+8 for upper lanes)

    // ---- GEMM1: [64,256] x [256,128] + rank-1 edge_attr update (fp32) ----
    float8 acc[4];
    {
        const float bb = b1[n];
        const float wl = w1_attr_row[n];
#pragma unroll
        for (int mt = 0; mt < 4; ++mt)
#pragma unroll
            for (int i = 0; i < 8; ++i)
                acc[mt][i] = bb + s_attr[mt * 16 + mSel + i] * wl;
    }
#pragma unroll
    for (int k = 0; k < 8; ++k) {
        const _Float16* src = (k < 4) ? s_hr : s_hc;
        const int kb = (k & 3) * 32;
        const half16 bf = load_b_sw(w1sw, k, nTile, lane);
#pragma unroll
        for (int mt = 0; mt < 4; ++mt) {
            const half16 af = load_a(src, mt * 16, kb, lane, HID);
            acc[mt] = wmma32(af, bf, acc[mt]);
        }
    }
#pragma unroll
    for (int mt = 0; mt < 4; ++mt)
#pragma unroll
        for (int i = 0; i < 8; ++i) {
            const int m = mt * 16 + mSel + i;
            s_mid[m * HID + n] = (_Float16)silu_f(acc[mt][i]);
        }
    __syncthreads();

    // ---- GEMM2: [64,128] x [128,128] ----
    float8 acc2[4];
    {
        const float bb = b2[n];
#pragma unroll
        for (int mt = 0; mt < 4; ++mt)
#pragma unroll
            for (int i = 0; i < 8; ++i) acc2[mt][i] = bb;
    }
#pragma unroll
    for (int k = 0; k < 4; ++k) {
        const half16 bf = load_b_sw(w2sw, k, nTile, lane);
#pragma unroll
        for (int mt = 0; mt < 4; ++mt) {
            const half16 af = load_a(s_mid, mt * 16, k * 32, lane, HID);
            acc2[mt] = wmma32(af, bf, acc2[mt]);
        }
    }
    // ---- SiLU + scatter-add (segment_sum over row) ----
#pragma unroll
    for (int mt = 0; mt < 4; ++mt)
#pragma unroll
        for (int i = 0; i < 8; ++i) {
            const int m = mt * 16 + mSel + i;
            const int r = s_r[m];
            if (r >= 0) {
                unsafeAtomicAdd(&mi[(size_t)r * HID + n], silu_f(acc2[mt][i]));
            }
        }
}

// Fused node update: h = silu(silu([h, m_i] @ msg_w1 + b1) @ msg_w2 + b2)
__global__ __launch_bounds__(256) void egnn_node_kernel(
    float* __restrict__ h, _Float16* __restrict__ hf,
    const float* __restrict__ mi,
    const _Float16* __restrict__ w1sw,  // swizzled [256 x 128] f16 fragments
    const float* __restrict__ b1,
    const _Float16* __restrict__ w2sw,  // swizzled [128 x 128] f16 fragments
    const float* __restrict__ b2, int N) {
    __shared__ __align__(16) _Float16 s_a[TILE_N * 2 * HID];   // [h | mi] as f16
    __shared__ __align__(16) _Float16 s_mid[TILE_N * HID];

    const int tid = threadIdx.x;
    const int n0 = blockIdx.x * TILE_N;

    for (int idx = tid; idx < TILE_N * (HID / 8); idx += 256) {
        const int nd = idx >> 4;
        const int ch = idx & 15;
        int g = n0 + nd; if (g >= N) g = N - 1;
        ((uint4*)s_a)[nd * 32 + ch] = ((const uint4*)(hf + (size_t)g * HID))[ch];
    }
    // Stage m_i (f32 -> f16), vectorized: float4 load, packed half4 store.
    for (int idx = tid; idx < TILE_N * (HID / 4); idx += 256) {
        const int nd = idx >> 5;                // HID/4 = 32 chunks per node
        const int ch = idx & 31;
        int g = n0 + nd; if (g >= N) g = N - 1;
        const float4 v = ((const float4*)(mi + (size_t)g * HID))[ch];
        half4 hv;
        hv[0] = (_Float16)v.x; hv[1] = (_Float16)v.y;
        hv[2] = (_Float16)v.z; hv[3] = (_Float16)v.w;
        *(half4*)(&s_a[nd * (2 * HID) + HID + (ch << 2)]) = hv;
    }
    __syncthreads();

    const int lane = tid & 31;
    const int nTile = tid >> 5;
    const int n = (nTile << 4) + (lane & 15);
    const int mSel = (lane >> 4) << 3;

    float8 acc[4];
    {
        const float bb = b1[n];
#pragma unroll
        for (int mt = 0; mt < 4; ++mt)
#pragma unroll
            for (int i = 0; i < 8; ++i) acc[mt][i] = bb;
    }
#pragma unroll
    for (int k = 0; k < 8; ++k) {
        const half16 bf = load_b_sw(w1sw, k, nTile, lane);
#pragma unroll
        for (int mt = 0; mt < 4; ++mt) {
            const half16 af = load_a(s_a, mt * 16, k * 32, lane, 2 * HID);
            acc[mt] = wmma32(af, bf, acc[mt]);
        }
    }
#pragma unroll
    for (int mt = 0; mt < 4; ++mt)
#pragma unroll
        for (int i = 0; i < 8; ++i) {
            const int m = mt * 16 + mSel + i;
            s_mid[m * HID + n] = (_Float16)silu_f(acc[mt][i]);
        }
    __syncthreads();

    float8 acc2[4];
    {
        const float bb = b2[n];
#pragma unroll
        for (int mt = 0; mt < 4; ++mt)
#pragma unroll
            for (int i = 0; i < 8; ++i) acc2[mt][i] = bb;
    }
#pragma unroll
    for (int k = 0; k < 4; ++k) {
        const half16 bf = load_b_sw(w2sw, k, nTile, lane);
#pragma unroll
        for (int mt = 0; mt < 4; ++mt) {
            const half16 af = load_a(s_mid, mt * 16, k * 32, lane, HID);
            acc2[mt] = wmma32(af, bf, acc2[mt]);
        }
    }
#pragma unroll
    for (int mt = 0; mt < 4; ++mt)
#pragma unroll
        for (int i = 0; i < 8; ++i) {
            const int m = mt * 16 + mSel + i;
            const int g = n0 + m;
            if (g < N) {
                const float v = silu_f(acc2[mt][i]);
                h[(size_t)g * HID + n] = v;
                hf[(size_t)g * HID + n] = (_Float16)v;
            }
        }
}

// out = silu(h @ out_w1 + b1) @ out_w2 + b2 ; one block (128 threads) per node.
__global__ __launch_bounds__(HID) void egnn_out_kernel(
    const float* __restrict__ h, const float* __restrict__ w1,
    const float* __restrict__ b1, const float* __restrict__ w2,
    const float* __restrict__ b2, float* __restrict__ out, int N) {
    const int nd = blockIdx.x;
    const int t = threadIdx.x;
    if (nd >= N) return;
    __shared__ float s_h[HID];
    __shared__ float s_t[HID];
    s_h[t] = h[(size_t)nd * HID + t];
    __syncthreads();
    float a = b1[t];
#pragma unroll 1
    for (int k = 0; k < HID; ++k) a += s_h[k] * w1[k * HID + t];
    s_t[t] = silu_f(a);
    __syncthreads();
    if (t < POS) {
        float o = b2[t];
#pragma unroll 1
        for (int k = 0; k < HID; ++k) o += s_t[k] * w2[k * POS + t];
        out[(size_t)nd * POS + t] = o;
    }
}

extern "C" void kernel_launch(void* const* d_in, const int* in_sizes, int n_in,
                              void* d_out, int out_size, void* d_ws, size_t ws_size,
                              hipStream_t stream) {
    const float* vel     = (const float*)d_in[0];
    const float* x       = (const float*)d_in[1];
    const int*   ei      = (const int*)d_in[2];
    const float* charges = (const float*)d_in[3];
    const float* ve_w    = (const float*)d_in[4];
    const float* ve_b    = (const float*)d_in[5];
    const float* node_w  = (const float*)d_in[6];
    const float* node_b  = (const float*)d_in[7];
    const float* edge_w1 = (const float*)d_in[8];
    const float* edge_b1 = (const float*)d_in[9];
    const float* edge_w2 = (const float*)d_in[10];
    const float* edge_b2 = (const float*)d_in[11];
    const float* msg_w1  = (const float*)d_in[12];
    const float* msg_b1  = (const float*)d_in[13];
    const float* msg_w2  = (const float*)d_in[14];
    const float* msg_b2  = (const float*)d_in[15];
    const float* out_w1  = (const float*)d_in[16];
    const float* out_b1  = (const float*)d_in[17];
    const float* out_w2  = (const float*)d_in[18];
    const float* out_b2  = (const float*)d_in[19];

    const int N = in_sizes[0];          // vel_norms [N,1]
    const int E = in_sizes[2] / 2;      // edge_index [2,E]
    const int L = in_sizes[9] / HID;    // edge_b1 [L,HID]

    char* ws = (char*)d_ws;
    size_t off = 0;
    auto carve = [&](size_t bytes) -> char* {
        char* p = ws + off;
        off += (bytes + 255) & ~(size_t)255;
        return p;
    };
    float*    h    = (float*)carve((size_t)N * HID * sizeof(float));
    _Float16* hf   = (_Float16*)carve((size_t)N * HID * sizeof(_Float16));
    float*    mi   = (float*)carve((size_t)N * HID * sizeof(float));
    _Float16* ew1s = (_Float16*)carve((size_t)L * 256 * HID * sizeof(_Float16));
    _Float16* ew2s = (_Float16*)carve((size_t)L * HID * HID * sizeof(_Float16));
    _Float16* mw1s = (_Float16*)carve((size_t)L * 2 * HID * HID * sizeof(_Float16));
    _Float16* mw2s = (_Float16*)carve((size_t)L * HID * HID * sizeof(_Float16));
    (void)ws_size; (void)n_in; (void)out_size;

    // Swizzle-convert all weights (per layer; edge_w1 uses rows 0..255, row 256 stays fp32).
    auto swz = [&](const float* s, _Float16* d, int K) {
        const int cnt = K * HID;
        swizzle_b_kernel<<<(cnt + 255) / 256, 256, 0, stream>>>(s, d, cnt);
    };
    for (int l = 0; l < L; ++l) {
        swz(edge_w1 + (size_t)l * 257 * HID, ew1s + (size_t)l * 256 * HID, 256);
        swz(edge_w2 + (size_t)l * HID * HID, ew2s + (size_t)l * HID * HID, HID);
        swz(msg_w1 + (size_t)l * 2 * HID * HID, mw1s + (size_t)l * 2 * HID * HID, 256);
        swz(msg_w2 + (size_t)l * HID * HID, mw2s + (size_t)l * HID * HID, HID);
    }

    egnn_encoder_kernel<<<N, HID, 0, stream>>>(vel, x, ve_w, ve_b, node_w, node_b,
                                               h, hf, N);

    const int hidElems = N * HID;
    for (int l = 0; l < L; ++l) {
        zero_kernel<<<(hidElems + 255) / 256, 256, 0, stream>>>(mi, hidElems);
        egnn_edge_kernel<<<(E + TILE_E - 1) / TILE_E, 256, 0, stream>>>(
            hf, ei, charges,
            ew1s + (size_t)l * 256 * HID,
            edge_w1 + (size_t)l * 257 * HID + (size_t)256 * HID,
            edge_b1 + (size_t)l * HID,
            ew2s + (size_t)l * HID * HID,
            edge_b2 + (size_t)l * HID,
            mi, E);
        egnn_node_kernel<<<(N + TILE_N - 1) / TILE_N, 256, 0, stream>>>(
            h, hf, mi,
            mw1s + (size_t)l * 2 * HID * HID,
            msg_b1 + (size_t)l * HID,
            mw2s + (size_t)l * HID * HID,
            msg_b2 + (size_t)l * HID,
            N);
    }

    egnn_out_kernel<<<N, HID, 0, stream>>>(h, out_w1, out_b1, out_w2, out_b2,
                                           (float*)d_out, N);
}
